// Apriel2GatedDeltaNet_40278203302193
// MI455X (gfx1250) — compile-verified, matrix operand order checked by reading, not run on hardware
//
#include <hip/hip_runtime.h>
#include <math.h>

// ---------------- problem constants ----------------
constexpr int B_   = 2;
constexpr int L_   = 4096;
constexpr int D_   = 2048;
constexpr int HK   = 8;
constexpr int HV   = 32;
constexpr int DK   = 64;
constexpr int DV   = 64;
constexpr int KDIM = HK * DK;             // 512
constexpr int VDIM = HV * DV;             // 2048
constexpr int CONV_DIM = 2 * KDIM + VDIM; // 3072
constexpr int KCONV = 4;
constexpr int QKVZ_N = 2 * KDIM + 2 * VDIM; // 5120
constexpr int BA_N   = 2 * HV;              // 64
constexpr int BL     = B_ * L_;             // 8192

// ---------------- feature detection ----------------
#ifndef __has_builtin
#define __has_builtin(x) 0
#endif

#if __has_builtin(__builtin_amdgcn_global_load_async_to_lds_b128)
#define HAVE_ASYNC_LDS 1
#else
#define HAVE_ASYNC_LDS 0
#endif

typedef int i32x4 __attribute__((vector_size(16)));   // matches builtin's pointee type

#if HAVE_ASYNC_LDS
#if __has_builtin(__builtin_amdgcn_s_wait_asynccnt)
#define WAIT_ASYNC(n) __builtin_amdgcn_s_wait_asynccnt(n)
#else
#define WAIT_ASYNC(n) asm volatile("s_wait_asynccnt %0" ::"i"(n) : "memory")
#endif
// global (AS1) and LDS (AS3) typed pointers, const dropped by C-style cast
#define ASYNC_CP_B128(g, l)                                          \
    __builtin_amdgcn_global_load_async_to_lds_b128(                  \
        (__attribute__((address_space(1))) i32x4*)(g),               \
        (__attribute__((address_space(3))) i32x4*)(l), 0, 0)
#else
#define WAIT_ASYNC(n)
#endif

// ---------------- WMMA types ----------------
typedef __attribute__((ext_vector_type(16))) __bf16 v16bf;
typedef __attribute__((ext_vector_type(8)))  float  v8f;

union FragBF {
    v16bf        v;
    unsigned int w[8];
    uint4        q[2];
};

__device__ __forceinline__ unsigned short f2bf(float f) {
    unsigned int u = __float_as_uint(f);
    unsigned int r = u + 0x7FFFu + ((u >> 16) & 1u);   // round-to-nearest-even
    return (unsigned short)(r >> 16);
}
__device__ __forceinline__ unsigned int pk2bf(float lo, float hi) {
    return (unsigned int)f2bf(lo) | ((unsigned int)f2bf(hi) << 16);
}
__device__ __forceinline__ int imin(int a, int b) { return a < b ? a : b; }

// =====================================================================
// Elementwise f32 -> bf16 (row-major preserved). n must be multiple of 8.
// =====================================================================
__global__ __launch_bounds__(256) void cvt_bf16_kernel(
    const float* __restrict__ in, unsigned short* __restrict__ out, size_t n)
{
    size_t i = ((size_t)blockIdx.x * 256 + threadIdx.x) * 8;
    if (i >= n) return;
    float4 a = *(const float4*)(in + i);
    float4 b = *(const float4*)(in + i + 4);
    uint4  o;
    o.x = pk2bf(a.x, a.y);
    o.y = pk2bf(a.z, a.w);
    o.z = pk2bf(b.x, b.y);
    o.w = pk2bf(b.z, b.w);
    *(uint4*)(out + i) = o;
}

// =====================================================================
// Transpose + convert: in f32 [K][N] -> out bf16 [N][K] (weights).
// =====================================================================
__global__ __launch_bounds__(256) void transpose_cvt_kernel(
    const float* __restrict__ in, unsigned short* __restrict__ out, int K, int N)
{
    __shared__ float tile[32][33];
    const int k0 = blockIdx.y * 32, n0 = blockIdx.x * 32;
    const int tx = threadIdx.x & 31, ty = threadIdx.x >> 5;   // 32 x 8
#pragma unroll
    for (int i = 0; i < 32; i += 8) {
        int k = k0 + ty + i, n = n0 + tx;
        tile[ty + i][tx] = (k < K && n < N) ? in[(size_t)k * N + n] : 0.0f;
    }
    __syncthreads();
#pragma unroll
    for (int i = 0; i < 32; i += 8) {
        int n = n0 + ty + i, k = k0 + tx;
        if (n < N && k < K) out[(size_t)n * K + k] = f2bf(tile[tx][ty + i]);
    }
}

// =====================================================================
// GEMM: C[M,N] = A[M,K] * Bt[N,K]^T, bf16 inputs, f32 WMMA accumulate.
// Block = 256 threads (8 wave32), tile 128x128, K-step 32.
// Double-buffered LDS, async global->LDS copies when available.
// =====================================================================
__global__ __launch_bounds__(256) void gemm_bf16_wmma(
    const unsigned short* __restrict__ A,   // [M][K] bf16
    const unsigned short* __restrict__ Bt,  // [N][K] bf16 (pre-transposed)
    float* __restrict__ C, int M, int N, int K)
{
    constexpr int BM = 128, BN = 128, BK = 32, PAD = 8;
    __shared__ __align__(16) unsigned short As[2][BM][BK + PAD];
    __shared__ __align__(16) unsigned short Bs[2][BN][BK + PAD];

    const int tid  = threadIdx.x;
    const int wid  = tid >> 5;
    const int lane = tid & 31;
    const int lm   = lane & 15;
    const int lh   = lane >> 4;
    const int mW   = (wid >> 2) * 64;
    const int nW   = (wid &  3) * 32;

    const int mBase = blockIdx.y * BM;
    const int nBase = blockIdx.x * BN;

    // copy map: thread owns 16 bf16 (32B) of one A row and one B row
    const int r_ = tid >> 1, c_ = (tid & 1) * 16;
    const int gm = imin(mBase + r_, M - 1);   // clamp OOB: uniform issue count,
    const int gn = imin(nBase + r_, N - 1);   // junk columns never stored

    auto copy_tile = [&](int buf, int kt) {
        const unsigned short* ga = A  + (size_t)gm * K + kt + c_;
        const unsigned short* gb = Bt + (size_t)gn * K + kt + c_;
        unsigned short* la = &As[buf][r_][c_];
        unsigned short* lb = &Bs[buf][r_][c_];
#if HAVE_ASYNC_LDS
        ASYNC_CP_B128(ga,     la);
        ASYNC_CP_B128(ga + 8, la + 8);
        ASYNC_CP_B128(gb,     lb);
        ASYNC_CP_B128(gb + 8, lb + 8);
#else
        *(uint4*)la       = *(const uint4*)ga;
        *(uint4*)(la + 8) = *(const uint4*)(ga + 8);
        *(uint4*)lb       = *(const uint4*)gb;
        *(uint4*)(lb + 8) = *(const uint4*)(gb + 8);
#endif
    };

    v8f acc[4][2] = {};

    copy_tile(0, 0);
    const int nK = K / BK;
    for (int kt = 0; kt < nK; ++kt) {
        const int cur = kt & 1;
        if (kt + 1 < nK) {
            copy_tile(cur ^ 1, (kt + 1) * BK);
            WAIT_ASYNC(4);   // previous tile done; newest 4 still in flight
        } else {
            WAIT_ASYNC(0);
        }
        if (kt + 2 < nK) {   // gfx1250 global_prefetch_b8 two tiles ahead
            __builtin_prefetch(A  + (size_t)gm * K + (kt + 2) * BK + c_, 0, 0);
            __builtin_prefetch(Bt + (size_t)gn * K + (kt + 2) * BK + c_, 0, 0);
        }
        __syncthreads();

        // fragments: two contiguous 16B K-runs per lane (ISA 16-bit layout)
        FragBF afrag[4], bfrag[2];
#pragma unroll
        for (int mt = 0; mt < 4; ++mt) {
            const unsigned short* row = &As[cur][mW + mt * 16 + lm][0];
            afrag[mt].q[0] = *(const uint4*)(row + 8 * lh);
            afrag[mt].q[1] = *(const uint4*)(row + 16 + 8 * lh);
        }
#pragma unroll
        for (int nt = 0; nt < 2; ++nt) {
            const unsigned short* col = &Bs[cur][nW + nt * 16 + lm][0];
            bfrag[nt].q[0] = *(const uint4*)(col + 8 * lh);
            bfrag[nt].q[1] = *(const uint4*)(col + 16 + 8 * lh);
        }

#pragma unroll
        for (int mt = 0; mt < 4; ++mt)
#pragma unroll
            for (int nt = 0; nt < 2; ++nt)
                acc[mt][nt] = __builtin_amdgcn_wmma_f32_16x16x32_bf16(
                    false, afrag[mt].v, false, bfrag[nt].v,
                    (short)0, acc[mt][nt], false, false);

        __syncthreads();
    }

#pragma unroll
    for (int mt = 0; mt < 4; ++mt)
#pragma unroll
        for (int nt = 0; nt < 2; ++nt)
#pragma unroll
            for (int r = 0; r < 8; ++r) {
                int m = mBase + mW + mt * 16 + r + 8 * lh;
                int n = nBase + nW + nt * 16 + lm;
                if (m < M && n < N) C[(size_t)m * N + n] = acc[mt][nt][r];
            }
}

// =====================================================================
// Causal depthwise conv(K=4) + SiLU + per-head l2norm + gates.
// =====================================================================
__global__ __launch_bounds__(256) void conv_gate_kernel(
    const float* __restrict__ qkvz, const float* __restrict__ ba,
    const float* __restrict__ conv_w, const float* __restrict__ dt_bias,
    const float* __restrict__ A_log,
    float* __restrict__ qn, float* __restrict__ kn, float* __restrict__ vs,
    float* __restrict__ g, float* __restrict__ beta)
{
    const int bl = blockIdx.x;
    const int b  = bl / L_;
    const int l  = bl % L_;
    const int tid = threadIdx.x;

    __shared__ float qkbuf[2 * KDIM];
    __shared__ float red[256];

    for (int c = tid; c < CONV_DIM; c += 256) {
        float acc = 0.0f;
#pragma unroll
        for (int j = 0; j < KCONV; ++j) {
            int lj = l - (KCONV - 1) + j;
            float x = (lj >= 0) ? qkvz[((size_t)(b * L_ + lj)) * QKVZ_N + c] : 0.0f;
            acc += conv_w[c * KCONV + j] * x;
        }
        float s = acc / (1.0f + __expf(-acc));   // SiLU
        if (c < 2 * KDIM) qkbuf[c] = s;
        else vs[(size_t)bl * VDIM + (c - 2 * KDIM)] = s;
    }
    __syncthreads();

    for (int it = 0; it < 4; ++it) {
        const int h  = it * 4 + (tid >> 6);
        const int li = tid & 63;
        float val = qkbuf[h * 64 + li];
        red[tid] = val * val;
        __syncthreads();
        for (int s = 32; s > 0; s >>= 1) {
            if (li < s) red[tid] += red[tid + s];
            __syncthreads();
        }
        float rn   = rsqrtf(red[tid & ~63] + 1e-6f);
        float outv = val * rn;
        if (h < HK) qn[(size_t)bl * KDIM + h * 64 + li] = outv * 0.125f; // * DK^-0.5
        else        kn[(size_t)bl * KDIM + (h - HK) * 64 + li] = outv;
        __syncthreads();
    }

    if (tid < HV) {
        float bg = ba[(size_t)bl * BA_N + tid];
        float ag = ba[(size_t)bl * BA_N + HV + tid];
        float x  = ag + dt_bias[tid];
        float sp = (x > 20.0f) ? x : log1pf(__expf(x));   // softplus
        g[(size_t)bl * HV + tid]    = -__expf(A_log[tid]) * sp;
        beta[(size_t)bl * HV + tid] = 1.0f / (1.0f + __expf(-bg));
    }
}

// =====================================================================
// Gated delta-rule recurrence: lane owns one V-column of S in registers.
// =====================================================================
__global__ __launch_bounds__(64) void scan_kernel(
    const float* __restrict__ qn, const float* __restrict__ kn,
    const float* __restrict__ vs, const float* __restrict__ g,
    const float* __restrict__ beta, float* __restrict__ o)
{
    const int bh = blockIdx.x;
    const int b  = bh / HV;
    const int h  = bh % HV;
    const int hk = h >> 2;            // GQA: HV/HK = 4
    const int v  = threadIdx.x;

    float S[64];
#pragma unroll
    for (int i = 0; i < 64; ++i) S[i] = 0.0f;

    __shared__ float kq[128];

    for (int t = 0; t < L_; ++t) {
        const size_t bl = (size_t)b * L_ + t;
        kq[v]      = kn[bl * KDIM + hk * 64 + v];
        kq[64 + v] = qn[bl * KDIM + hk * 64 + v];
        const float vt = vs[bl * VDIM + h * 64 + v];
        const float gt = g[bl * HV + h];
        const float bt = beta[bl * HV + h];
        __syncthreads();

        const float eg = __expf(gt);
        float kv = 0.0f;
#pragma unroll
        for (int kk = 0; kk < 64; ++kk) {
            S[kk] *= eg;
            kv += kq[kk] * S[kk];
        }
        const float dl = (vt - kv) * bt;
        float ot = 0.0f;
#pragma unroll
        for (int kk = 0; kk < 64; ++kk) {
            S[kk] += kq[kk] * dl;
            ot += kq[64 + kk] * S[kk];
        }
        o[bl * VDIM + h * 64 + v] = ot;
        __syncthreads();
    }
}

// =====================================================================
// Gated RMSNorm per head (in-place on o).
// =====================================================================
__global__ __launch_bounds__(64) void norm_kernel(
    const float* __restrict__ qkvz, const float* __restrict__ nw,
    float* __restrict__ o)
{
    const size_t idx = blockIdx.x;
    const size_t bl  = idx / HV;
    const int    h   = (int)(idx % HV);
    const int    v   = threadIdx.x;

    const float z  = qkvz[bl * QKVZ_N + (2 * KDIM + VDIM) + h * 64 + v];
    const float ov = o[bl * VDIM + h * 64 + v];
    const float x  = ov * (z / (1.0f + __expf(-z)));

    __shared__ float red[64];
    red[v] = x * x;
    __syncthreads();
    for (int s = 32; s > 0; s >>= 1) {
        if (v < s) red[v] += red[v + s];
        __syncthreads();
    }
    const float var = red[0] * (1.0f / 64.0f);
    o[bl * VDIM + h * 64 + v] = x * rsqrtf(var + 1e-5f) * nw[v];
}

// =====================================================================
extern "C" void kernel_launch(void* const* d_in, const int* in_sizes, int n_in,
                              void* d_out, int out_size, void* d_ws, size_t ws_size,
                              hipStream_t stream) {
    const float* hidden   = (const float*)d_in[0];  // [B,L,D]
    const float* W_qkvz   = (const float*)d_in[1];  // [D, 5120]
    const float* W_ba     = (const float*)d_in[2];  // [D, 64]
    const float* conv_w   = (const float*)d_in[3];  // [3072,1,4]
    const float* dt_bias  = (const float*)d_in[4];  // [HV]
    const float* A_log    = (const float*)d_in[5];  // [HV]
    const float* norm_w   = (const float*)d_in[6];  // [DV]
    const float* W_out    = (const float*)d_in[7];  // [VDIM, D]
    float*       out      = (float*)d_out;          // [B,L,D]

    // ---------- workspace layout ----------
    float* ws   = (float*)d_ws;
    float* qkvz = ws;                              // BL*5120 f32
    float* baP  = qkvz + (size_t)BL * QKVZ_N;      // BL*64
    float* qn   = baP  + (size_t)BL * BA_N;        // BL*512
    float* kn   = qn   + (size_t)BL * KDIM;        // BL*512
    float* vsP  = kn   + (size_t)BL * KDIM;        // BL*2048
    float* gP   = vsP  + (size_t)BL * VDIM;        // BL*32
    float* btP  = gP   + (size_t)BL * HV;          // BL*32
    float* oP   = btP  + (size_t)BL * HV;          // BL*2048
    unsigned short* hidBf = (unsigned short*)(oP + (size_t)BL * VDIM); // BL*2048 bf16
    unsigned short* WqT   = hidBf + (size_t)BL * D_;      // [5120][2048] bf16
    unsigned short* WbaT  = WqT   + (size_t)QKVZ_N * D_;  // [64][2048]
    unsigned short* WoT   = WbaT  + (size_t)BA_N * D_;    // [2048][2048]
    unsigned short* xBf   = WoT   + (size_t)D_ * VDIM;    // BL*2048 bf16

    dim3 blk(256);

    // 0) precision staging: bf16 activations, bf16-transposed weights
    {
        size_t n = (size_t)BL * D_;
        cvt_bf16_kernel<<<dim3((unsigned)((n / 8 + 255) / 256)), blk, 0, stream>>>(hidden, hidBf, n);
    }
    transpose_cvt_kernel<<<dim3(QKVZ_N / 32, D_ / 32), blk, 0, stream>>>(W_qkvz, WqT, D_, QKVZ_N);
    transpose_cvt_kernel<<<dim3(BA_N / 32,   D_ / 32), blk, 0, stream>>>(W_ba,   WbaT, D_, BA_N);
    transpose_cvt_kernel<<<dim3(D_ / 32,     VDIM / 32), blk, 0, stream>>>(W_out, WoT, VDIM, D_);

    // 1) qkvz = hidden @ W_qkvz   [8192,2048]x[2048,5120]
    gemm_bf16_wmma<<<dim3(QKVZ_N / 128, BL / 128), blk, 0, stream>>>(
        hidBf, WqT, qkvz, BL, QKVZ_N, D_);

    // 2) ba = hidden @ W_ba       [8192,2048]x[2048,64]
    gemm_bf16_wmma<<<dim3(1, BL / 128), blk, 0, stream>>>(
        hidBf, WbaT, baP, BL, BA_N, D_);

    // 3) conv + SiLU + l2norm + gates
    conv_gate_kernel<<<dim3(BL), blk, 0, stream>>>(
        qkvz, baP, conv_w, dt_bias, A_log, qn, kn, vsP, gP, btP);

    // 4) gated delta-rule scan
    scan_kernel<<<dim3(B_ * HV), dim3(64), 0, stream>>>(qn, kn, vsP, gP, btP, oP);

    // 5) gated RMSNorm (in-place on o)
    norm_kernel<<<dim3(B_ * L_ * HV), dim3(64), 0, stream>>>(qkvz, norm_w, oP);

    // 5b) convert normed activations to bf16
    {
        size_t n = (size_t)BL * VDIM;
        cvt_bf16_kernel<<<dim3((unsigned)((n / 8 + 255) / 256)), blk, 0, stream>>>(oP, xBf, n);
    }

    // 6) out = x @ W_out          [8192,2048]x[2048,2048]
    gemm_bf16_wmma<<<dim3(D_ / 128, BL / 128), blk, 0, stream>>>(
        xBf, WoT, out, BL, D_, VDIM);
}
